// GraphNet_18751827214707
// MI455X (gfx1250) — compile-verified, matrix operand order checked by reading, not run on hardware
//
#include <hip/hip_runtime.h>
#include <hip/hip_bf16.h>

#define N_CNT 50000
#define E_CNT 800000
#define G_CNT 64
#define LN_EPS 1e-5f

typedef __attribute__((ext_vector_type(16))) __bf16 v16bf;
typedef __attribute__((ext_vector_type(8)))  float  v8f;

union ABf {
    v16bf v;
    __bf16 b[16];
    unsigned short h[16];
    uint4 q[2];
};

// Native bf16 convert: clang lowers fptrunc f32->bf16 to the gfx1250
// hardware converts (v_cvt_pk_bf16_f32 when pairs combine).
__device__ __forceinline__ __bf16 f2bf16(float f) { return (__bf16)f; }

__device__ __forceinline__ unsigned short f2bfu(float f) {
    return __builtin_bit_cast(unsigned short, (__bf16)f);
}

__device__ __forceinline__ void atomAddF(float* p, float v) {
    unsafeAtomicAdd(p, v);   // hardware global_atomic_add_f32
}

// ---------------------------------------------------------------------------
// Weight prep: transpose to [N][K] and convert f32 -> bf16.
// Offsets (u16 elements) inside wT:
//   ew1T [128][224] @ 0        (28672)
//   ew2T [ 64][128] @ 28672    ( 8192)
//   nw1T [128][160] @ 36864    (20480)
//   nw2T [ 64][128] @ 57344    ( 8192)
//   gw1T [128][160] @ 65536    (20480)
//   gw2T [ 32][128] @ 86016    ( 4096)  total 90112
// ---------------------------------------------------------------------------
#define OFF_EW1T 0
#define OFF_EW2T 28672
#define OFF_NW1T 36864
#define OFF_NW2T 57344
#define OFF_GW1T 65536
#define OFF_GW2T 86016
#define WT_ELEMS 90112

__global__ __launch_bounds__(256) void prep_weights(
    const float* __restrict__ ew1, const float* __restrict__ ew2,
    const float* __restrict__ nw1, const float* __restrict__ nw2,
    const float* __restrict__ gw1, const float* __restrict__ gw2,
    unsigned short* __restrict__ wT)
{
    int t = blockIdx.x * 256 + threadIdx.x;
    if (t >= WT_ELEMS) return;
    int i = t;
    float v;
    if (i < 28672)              { int n = i / 224, k = i % 224; v = ew1[k * 128 + n]; }
    else if ((i -= 28672) < 8192)  { int n = i / 128, k = i % 128; v = ew2[k * 64 + n]; }
    else if ((i -= 8192) < 20480)  { int n = i / 160, k = i % 160; v = nw1[k * 128 + n]; }
    else if ((i -= 20480) < 8192)  { int n = i / 128, k = i % 128; v = nw2[k * 64 + n]; }
    else if ((i -= 8192) < 20480)  { int n = i / 160, k = i % 160; v = gw1[k * 128 + n]; }
    else { i -= 20480;             int n = i / 128, k = i % 128; v = gw2[k * 32 + n]; }
    wT[t] = f2bfu(v);
}

__global__ __launch_bounds__(256) void zero_f32(float* __restrict__ p, int n) {
    int i = blockIdx.x * 256 + threadIdx.x;
    if (i < n) p[i] = 0.f;
}

// ---------------------------------------------------------------------------
// Edge MLP: concat(x[row],x[col],edge_attr,u[e_idx]) (224) -> 128 -> 64, LN.
// 1 wave per 16-edge tile, 8 waves / block. Also scatter-adds e_new into agg.
// ---------------------------------------------------------------------------
__global__ __launch_bounds__(256) void edge_kernel(
    const float* __restrict__ x, const float* __restrict__ edge_attr,
    const float* __restrict__ u,
    const float* __restrict__ eb1, const float* __restrict__ eb2,
    const float* __restrict__ eg,  const float* __restrict__ ebt,
    const int* __restrict__ edge_index, const int* __restrict__ e_indices,
    const unsigned short* __restrict__ w1T,   // [128][224] bf16
    const unsigned short* __restrict__ w2T,   // [ 64][128] bf16
    float* __restrict__ e_new, float* __restrict__ agg)
{
    __shared__ alignas(16) unsigned short h1[8][16 * 128];
    __shared__ alignas(16) float           ln[8][16 * 64];

    const int wid  = threadIdx.x >> 5;
    const int lane = threadIdx.x & 31;
    const int M    = lane & 15;
    const int hi   = lane >> 4;
    const int tile = blockIdx.x * 8 + wid;      // E/16 = 50000 tiles, exact
    const int edge = tile * 16 + M;

    __builtin_prefetch(w1T, 0, 3);
    __builtin_prefetch(w2T, 0, 3);

    const int row  = edge_index[edge];
    const int col  = edge_index[E_CNT + edge];
    const int gidx = e_indices[edge];

    const float* srcs[7] = {
        x + (size_t)row * 64,       x + (size_t)row * 64 + 32,
        x + (size_t)col * 64,       x + (size_t)col * 64 + 32,
        edge_attr + (size_t)edge * 64, edge_attr + (size_t)edge * 64 + 32,
        u + (size_t)gidx * 32
    };

    // ---- layer 1: 224 -> 128 ----
    v8f acc[8] = {};
#pragma unroll
    for (int ks = 0; ks < 7; ++ks) {
        ABf a;
        const float* s0 = srcs[ks] + hi * 8;
        const float* s1 = srcs[ks] + 16 + hi * 8;
#pragma unroll
        for (int e2 = 0; e2 < 8; ++e2) { a.b[e2] = f2bf16(s0[e2]); a.b[8 + e2] = f2bf16(s1[e2]); }
#pragma unroll
        for (int nt = 0; nt < 8; ++nt) {
            ABf b;
            const unsigned short* wp = w1T + (size_t)(nt * 16 + M) * 224 + ks * 32 + hi * 8;
            b.q[0] = *reinterpret_cast<const uint4*>(wp);
            b.q[1] = *reinterpret_cast<const uint4*>(wp + 16);
            acc[nt] = __builtin_amdgcn_wmma_f32_16x16x32_bf16(
                false, a.v, false, b.v, (short)0, acc[nt], false, false);
        }
    }
    // bias + ReLU + restage as bf16 A-matrix (D layout -> LDS row major)
#pragma unroll
    for (int nt = 0; nt < 8; ++nt) {
        float bias = eb1[nt * 16 + M];
#pragma unroll
        for (int v2 = 0; v2 < 8; ++v2) {
            float val = acc[nt][v2] + bias;
            val = val > 0.f ? val : 0.f;
            h1[wid][(v2 + hi * 8) * 128 + nt * 16 + M] = f2bfu(val);
        }
    }

    // ---- layer 2: 128 -> 64 ----
    v8f acc2[4] = {};
#pragma unroll
    for (int ks = 0; ks < 4; ++ks) {
        ABf a;
        const unsigned short* hp = &h1[wid][M * 128 + ks * 32 + hi * 8];
        a.q[0] = *reinterpret_cast<const uint4*>(hp);
        a.q[1] = *reinterpret_cast<const uint4*>(hp + 16);
#pragma unroll
        for (int nt = 0; nt < 4; ++nt) {
            ABf b;
            const unsigned short* wp = w2T + (size_t)(nt * 16 + M) * 128 + ks * 32 + hi * 8;
            b.q[0] = *reinterpret_cast<const uint4*>(wp);
            b.q[1] = *reinterpret_cast<const uint4*>(wp + 16);
            acc2[nt] = __builtin_amdgcn_wmma_f32_16x16x32_bf16(
                false, a.v, false, b.v, (short)0, acc2[nt], false, false);
        }
    }
#pragma unroll
    for (int nt = 0; nt < 4; ++nt) {
        float bias = eb2[nt * 16 + M];
#pragma unroll
        for (int v2 = 0; v2 < 8; ++v2) {
            float val = acc2[nt][v2] + bias;
            val = val > 0.f ? val : 0.f;
            ln[wid][(v2 + hi * 8) * 64 + nt * 16 + M] = val;
        }
    }

    // ---- LayerNorm + output + scatter-add into agg[row] ----
    if (lane < 16) {
        const float* hrow = &ln[wid][lane * 64];
        float mu = 0.f;
        for (int n = 0; n < 64; ++n) mu += hrow[n];
        mu *= (1.f / 64.f);
        float var = 0.f;
        for (int n = 0; n < 64; ++n) { float d = hrow[n] - mu; var += d * d; }
        var *= (1.f / 64.f);
        float rs = rsqrtf(var + LN_EPS);
        float* outp = e_new + (size_t)edge * 64;
        float* aggp = agg + (size_t)row * 64;
        for (int n = 0; n < 64; ++n) {
            float o = (hrow[n] - mu) * rs * eg[n] + ebt[n];
            outp[n] = o;
            atomAddF(&aggp[n], o);
        }
    }
}

// ---------------------------------------------------------------------------
// Node MLP: concat(x,agg,u[v_idx]) (160) -> 128 -> 64, LN.
// ---------------------------------------------------------------------------
__global__ __launch_bounds__(256) void node_kernel(
    const float* __restrict__ x, const float* __restrict__ agg,
    const float* __restrict__ u,
    const float* __restrict__ nb1, const float* __restrict__ nb2,
    const float* __restrict__ ng,  const float* __restrict__ nbt,
    const int* __restrict__ v_indices,
    const unsigned short* __restrict__ w1T,   // [128][160]
    const unsigned short* __restrict__ w2T,   // [ 64][128]
    float* __restrict__ x_new)
{
    __shared__ alignas(16) unsigned short h1[8][16 * 128];
    __shared__ alignas(16) float           ln[8][16 * 64];

    const int wid  = threadIdx.x >> 5;
    const int lane = threadIdx.x & 31;
    const int M    = lane & 15;
    const int hi   = lane >> 4;
    const int tile = blockIdx.x * 8 + wid;
    if (tile >= N_CNT / 16) return;             // whole-wave exit: EXEC stays full
    const int node = tile * 16 + M;
    const int gidx = v_indices[node];

    const float* srcs[5] = {
        x   + (size_t)node * 64, x   + (size_t)node * 64 + 32,
        agg + (size_t)node * 64, agg + (size_t)node * 64 + 32,
        u + (size_t)gidx * 32
    };

    v8f acc[8] = {};
#pragma unroll
    for (int ks = 0; ks < 5; ++ks) {
        ABf a;
        const float* s0 = srcs[ks] + hi * 8;
        const float* s1 = srcs[ks] + 16 + hi * 8;
#pragma unroll
        for (int e2 = 0; e2 < 8; ++e2) { a.b[e2] = f2bf16(s0[e2]); a.b[8 + e2] = f2bf16(s1[e2]); }
#pragma unroll
        for (int nt = 0; nt < 8; ++nt) {
            ABf b;
            const unsigned short* wp = w1T + (size_t)(nt * 16 + M) * 160 + ks * 32 + hi * 8;
            b.q[0] = *reinterpret_cast<const uint4*>(wp);
            b.q[1] = *reinterpret_cast<const uint4*>(wp + 16);
            acc[nt] = __builtin_amdgcn_wmma_f32_16x16x32_bf16(
                false, a.v, false, b.v, (short)0, acc[nt], false, false);
        }
    }
#pragma unroll
    for (int nt = 0; nt < 8; ++nt) {
        float bias = nb1[nt * 16 + M];
#pragma unroll
        for (int v2 = 0; v2 < 8; ++v2) {
            float val = acc[nt][v2] + bias;
            val = val > 0.f ? val : 0.f;
            h1[wid][(v2 + hi * 8) * 128 + nt * 16 + M] = f2bfu(val);
        }
    }

    v8f acc2[4] = {};
#pragma unroll
    for (int ks = 0; ks < 4; ++ks) {
        ABf a;
        const unsigned short* hp = &h1[wid][M * 128 + ks * 32 + hi * 8];
        a.q[0] = *reinterpret_cast<const uint4*>(hp);
        a.q[1] = *reinterpret_cast<const uint4*>(hp + 16);
#pragma unroll
        for (int nt = 0; nt < 4; ++nt) {
            ABf b;
            const unsigned short* wp = w2T + (size_t)(nt * 16 + M) * 128 + ks * 32 + hi * 8;
            b.q[0] = *reinterpret_cast<const uint4*>(wp);
            b.q[1] = *reinterpret_cast<const uint4*>(wp + 16);
            acc2[nt] = __builtin_amdgcn_wmma_f32_16x16x32_bf16(
                false, a.v, false, b.v, (short)0, acc2[nt], false, false);
        }
    }
#pragma unroll
    for (int nt = 0; nt < 4; ++nt) {
        float bias = nb2[nt * 16 + M];
#pragma unroll
        for (int v2 = 0; v2 < 8; ++v2) {
            float val = acc2[nt][v2] + bias;
            val = val > 0.f ? val : 0.f;
            ln[wid][(v2 + hi * 8) * 64 + nt * 16 + M] = val;
        }
    }

    if (lane < 16) {
        const float* hrow = &ln[wid][lane * 64];
        float mu = 0.f;
        for (int n = 0; n < 64; ++n) mu += hrow[n];
        mu *= (1.f / 64.f);
        float var = 0.f;
        for (int n = 0; n < 64; ++n) { float d = hrow[n] - mu; var += d * d; }
        var *= (1.f / 64.f);
        float rs = rsqrtf(var + LN_EPS);
        float* outp = x_new + (size_t)node * 64;
        for (int n = 0; n < 64; ++n)
            outp[n] = (hrow[n] - mu) * rs * ng[n] + nbt[n];
    }
}

// ---------------------------------------------------------------------------
// Per-graph segment means of x_new (by v_indices) and e_new (by e_indices).
// Indices are sorted -> binary search for segment bounds.
// ---------------------------------------------------------------------------
__device__ __forceinline__ int lowerB(const int* __restrict__ a, int n, int key) {
    int lo = 0, hiB = n;
    while (lo < hiB) { int m = (lo + hiB) >> 1; if (a[m] < key) lo = m + 1; else hiB = m; }
    return lo;
}

__global__ __launch_bounds__(256) void graph_reduce(
    const float* __restrict__ xn, const float* __restrict__ en,
    const int* __restrict__ vind, const int* __restrict__ eind,
    float* __restrict__ meanX, float* __restrict__ meanE)
{
    __shared__ float red[256];
    const int g = blockIdx.x;
    const int t = threadIdx.x;
    const int f = t & 63, j = t >> 6;

    int vs = lowerB(vind, N_CNT, g), ve = lowerB(vind, N_CNT, g + 1);
    float s = 0.f;
    for (int r = vs + j; r < ve; r += 4) s += xn[(size_t)r * 64 + f];
    red[t] = s;
    __syncthreads();
    if (j == 0) {
        float tot = red[f] + red[f + 64] + red[f + 128] + red[f + 192];
        float c = (float)(ve - vs); if (c < 1.f) c = 1.f;
        meanX[g * 64 + f] = tot / c;
    }
    __syncthreads();

    int es = lowerB(eind, E_CNT, g), ee = lowerB(eind, E_CNT, g + 1);
    s = 0.f;
    for (int r = es + j; r < ee; r += 4) s += en[(size_t)r * 64 + f];
    red[t] = s;
    __syncthreads();
    if (j == 0) {
        float tot = red[f] + red[f + 64] + red[f + 128] + red[f + 192];
        float c = (float)(ee - es); if (c < 1.f) c = 1.f;
        meanE[g * 64 + f] = tot / c;
    }
}

// ---------------------------------------------------------------------------
// Global MLP: concat(u, mean_x, mean_e) (160) -> 128 -> 32, LN. 64 rows.
// ---------------------------------------------------------------------------
__global__ __launch_bounds__(128) void global_kernel(
    const float* __restrict__ u, const float* __restrict__ meanX,
    const float* __restrict__ meanE,
    const float* __restrict__ gb1, const float* __restrict__ gb2,
    const float* __restrict__ gg,  const float* __restrict__ gbt,
    const unsigned short* __restrict__ w1T,   // [128][160]
    const unsigned short* __restrict__ w2T,   // [ 32][128]
    float* __restrict__ u_new)
{
    __shared__ alignas(16) unsigned short h1[4][16 * 128];
    __shared__ alignas(16) float           ln[4][16 * 32];

    const int wid  = threadIdx.x >> 5;
    const int lane = threadIdx.x & 31;
    const int M    = lane & 15;
    const int hi   = lane >> 4;
    const int gr   = wid * 16 + M;     // graph row, 4 waves * 16 = 64

    const float* srcs[5] = {
        u + (size_t)gr * 32,
        meanX + (size_t)gr * 64, meanX + (size_t)gr * 64 + 32,
        meanE + (size_t)gr * 64, meanE + (size_t)gr * 64 + 32
    };

    v8f acc[8] = {};
#pragma unroll
    for (int ks = 0; ks < 5; ++ks) {
        ABf a;
        const float* s0 = srcs[ks] + hi * 8;
        const float* s1 = srcs[ks] + 16 + hi * 8;
#pragma unroll
        for (int e2 = 0; e2 < 8; ++e2) { a.b[e2] = f2bf16(s0[e2]); a.b[8 + e2] = f2bf16(s1[e2]); }
#pragma unroll
        for (int nt = 0; nt < 8; ++nt) {
            ABf b;
            const unsigned short* wp = w1T + (size_t)(nt * 16 + M) * 160 + ks * 32 + hi * 8;
            b.q[0] = *reinterpret_cast<const uint4*>(wp);
            b.q[1] = *reinterpret_cast<const uint4*>(wp + 16);
            acc[nt] = __builtin_amdgcn_wmma_f32_16x16x32_bf16(
                false, a.v, false, b.v, (short)0, acc[nt], false, false);
        }
    }
#pragma unroll
    for (int nt = 0; nt < 8; ++nt) {
        float bias = gb1[nt * 16 + M];
#pragma unroll
        for (int v2 = 0; v2 < 8; ++v2) {
            float val = acc[nt][v2] + bias;
            val = val > 0.f ? val : 0.f;
            h1[wid][(v2 + hi * 8) * 128 + nt * 16 + M] = f2bfu(val);
        }
    }

    v8f acc2[2] = {};
#pragma unroll
    for (int ks = 0; ks < 4; ++ks) {
        ABf a;
        const unsigned short* hp = &h1[wid][M * 128 + ks * 32 + hi * 8];
        a.q[0] = *reinterpret_cast<const uint4*>(hp);
        a.q[1] = *reinterpret_cast<const uint4*>(hp + 16);
#pragma unroll
        for (int nt = 0; nt < 2; ++nt) {
            ABf b;
            const unsigned short* wp = w2T + (size_t)(nt * 16 + M) * 128 + ks * 32 + hi * 8;
            b.q[0] = *reinterpret_cast<const uint4*>(wp);
            b.q[1] = *reinterpret_cast<const uint4*>(wp + 16);
            acc2[nt] = __builtin_amdgcn_wmma_f32_16x16x32_bf16(
                false, a.v, false, b.v, (short)0, acc2[nt], false, false);
        }
    }
#pragma unroll
    for (int nt = 0; nt < 2; ++nt) {
        float bias = gb2[nt * 16 + M];
#pragma unroll
        for (int v2 = 0; v2 < 8; ++v2) {
            float val = acc2[nt][v2] + bias;
            val = val > 0.f ? val : 0.f;
            ln[wid][(v2 + hi * 8) * 32 + nt * 16 + M] = val;
        }
    }

    if (lane < 16) {
        const float* hrow = &ln[wid][lane * 32];
        float mu = 0.f;
        for (int n = 0; n < 32; ++n) mu += hrow[n];
        mu *= (1.f / 32.f);
        float var = 0.f;
        for (int n = 0; n < 32; ++n) { float d = hrow[n] - mu; var += d * d; }
        var *= (1.f / 32.f);
        float rs = rsqrtf(var + LN_EPS);
        float* outp = u_new + (size_t)(wid * 16 + lane) * 32;
        for (int n = 0; n < 32; ++n)
            outp[n] = (hrow[n] - mu) * rs * gg[n] + gbt[n];
    }
}

// ---------------------------------------------------------------------------
extern "C" void kernel_launch(void* const* d_in, const int* in_sizes, int n_in,
                              void* d_out, int out_size, void* d_ws, size_t ws_size,
                              hipStream_t stream) {
    const float* x         = (const float*)d_in[0];
    const float* edge_attr = (const float*)d_in[1];
    const float* u         = (const float*)d_in[2];
    const float* ew1 = (const float*)d_in[3];
    const float* eb1 = (const float*)d_in[4];
    const float* ew2 = (const float*)d_in[5];
    const float* eb2 = (const float*)d_in[6];
    const float* eg  = (const float*)d_in[7];
    const float* ebt = (const float*)d_in[8];
    const float* nw1 = (const float*)d_in[9];
    const float* nb1 = (const float*)d_in[10];
    const float* nw2 = (const float*)d_in[11];
    const float* nb2 = (const float*)d_in[12];
    const float* ng  = (const float*)d_in[13];
    const float* nbt = (const float*)d_in[14];
    const float* gw1 = (const float*)d_in[15];
    const float* gb1 = (const float*)d_in[16];
    const float* gw2 = (const float*)d_in[17];
    const float* gb2 = (const float*)d_in[18];
    const float* gg  = (const float*)d_in[19];
    const float* gbt = (const float*)d_in[20];
    const int* edge_index = (const int*)d_in[21];
    const int* v_indices  = (const int*)d_in[22];
    const int* e_indices  = (const int*)d_in[23];

    // workspace layout
    const size_t AGG_BYTES = (size_t)N_CNT * 64 * sizeof(float);   // 12.8 MB
    float* agg = (float*)d_ws;
    unsigned short* wT = (unsigned short*)((char*)d_ws + AGG_BYTES);
    float* meanX = (float*)((char*)d_ws + AGG_BYTES + (size_t)WT_ELEMS * 2);
    float* meanE = meanX + G_CNT * 64;

    // output layout: x_new (N*64), e_new (E*64), u_new (G*32)
    float* x_new = (float*)d_out;
    float* e_new = x_new + (size_t)N_CNT * 64;
    float* u_new = e_new + (size_t)E_CNT * 64;

    prep_weights<<<(WT_ELEMS + 255) / 256, 256, 0, stream>>>(ew1, ew2, nw1, nw2, gw1, gw2, wT);
    zero_f32<<<(N_CNT * 64) / 256, 256, 0, stream>>>(agg, N_CNT * 64);
    edge_kernel<<<E_CNT / (16 * 8), 256, 0, stream>>>(
        x, edge_attr, u, eb1, eb2, eg, ebt, edge_index, e_indices,
        wT + OFF_EW1T, wT + OFF_EW2T, e_new, agg);
    node_kernel<<<(N_CNT / 16 + 7) / 8, 256, 0, stream>>>(
        x, agg, u, nb1, nb2, ng, nbt, v_indices,
        wT + OFF_NW1T, wT + OFF_NW2T, x_new);
    graph_reduce<<<G_CNT, 256, 0, stream>>>(x_new, e_new, v_indices, e_indices, meanX, meanE);
    global_kernel<<<1, 128, 0, stream>>>(
        u, meanX, meanE, gb1, gb2, gg, gbt,
        wT + OFF_GW1T, wT + OFF_GW2T, u_new);
}
